// RGATSQL_89593017795077
// MI455X (gfx1250) — matrix operand order repaired and checked by reading, not compiled, not run on hardware
//
#include <hip/hip_runtime.h>

typedef __attribute__((ext_vector_type(16))) __bf16 v16bf;
typedef __attribute__((ext_vector_type(8)))  float  v8f;

#define N_NODES 20000
#define N_EDGES 160000
#define DIM     512
#define HEADS   8
#define DKH     64
#define LAYERS  2
#define LN_EPS  1e-5f

#define WMMA_BF16(a, b, c) \
    __builtin_amdgcn_wmma_f32_16x16x32_bf16(false, (a), false, (b), (short)0, (c), false, false)

union Frag { uint4 u[2]; v16bf v; };

// LDS tile geometry (bytes)
#define A_ROW_STRIDE 80              // 64B of data + 16B pad (bank spread)
#define A_TILE_BYTES (128 * A_ROW_STRIDE)   // 10240
#define B_TILE_BYTES 8192            // 8 ntiles x 32 lanes x 32B
#define BUF_BYTES    (A_TILE_BYTES + B_TILE_BYTES)  // 18432

// ---------------------------------------------------------------------------
// f32 -> bf16 (activations)
// ---------------------------------------------------------------------------
__global__ __launch_bounds__(256)
void f2bf_kernel(const float* __restrict__ in, __bf16* __restrict__ out, int n) {
    int t = blockIdx.x * 256 + threadIdx.x;
    if (t < n) out[t] = (__bf16)in[t];
}

// ---------------------------------------------------------------------------
// Pack f32 weight W[K x Nc] into bf16 WMMA-B fragment layout:
//   out[(((nt*ktiles + kt)*32 + lane)*16 + j)] = W[kt*32 + (lane>>4)*16 + j][nt*16 + (lane&15)]
// so each lane's 16-element B fragment is one contiguous 32-byte chunk.
// ---------------------------------------------------------------------------
__global__ __launch_bounds__(256)
void pack_b_kernel(const float* __restrict__ W, __bf16* __restrict__ out, int K, int Nc) {
    int idx = blockIdx.x * 256 + threadIdx.x;
    if (idx >= K * Nc) return;
    const int j      = idx & 15;
    const int lane   = (idx >> 4) & 31;
    const int tile   = idx >> 9;
    const int ktiles = K >> 5;
    const int kt     = tile % ktiles;
    const int nt     = tile / ktiles;
    const int row    = (kt << 5) + ((lane >> 4) << 4) + j;
    const int col    = (nt << 4) + (lane & 15);
    out[idx] = (__bf16)W[(size_t)row * Nc + col];
}

// ---------------------------------------------------------------------------
// WMMA bf16 GEMM: C[M x Nc] = A[M x K] @ B[K x Nc] (+bias) (+relu)
// A bf16 row-major; Bp pre-packed (pack_b_kernel layout).
// Block = 256 threads = 8 waves (4 M-groups x 2 N-groups): block tile 128x128.
// LDS-staged, double-buffered: per 32-deep K step the block cooperatively
// stages the 128x32 A tile + 8 packed B tiles (16KB data) into LDS; the
// global loads for step k+1 issue right after the ds_store of step k, so
// global latency is hidden behind 8 WMMAs + short-latency ds_load fragments.
// ---------------------------------------------------------------------------
template<int K>
__global__ __launch_bounds__(256)
void gemm_wmma_bf16_t(const __bf16* __restrict__ A, const __bf16* __restrict__ Bp,
                      const float* __restrict__ bias,
                      float* __restrict__ Cf, __bf16* __restrict__ Cb,
                      int M, int Nc, int relu)
{
    constexpr int KT = K >> 5;               // number of 32-deep K tiles
    __shared__ __align__(16) unsigned char lds[2][BUF_BYTES];

    const int td     = threadIdx.x;
    const int lane   = td & 31;
    const int wave   = td >> 5;
    const int wm     = wave & 3;             // 0..3  -> 32-row group
    const int wn     = wave >> 2;            // 0..1  -> 64-col group
    const int lh     = lane & 15;
    const int khalfA = (lane >> 4) << 4;     // 0 or 16 (contiguous 16-bit A layout)

    const int rowBase = blockIdx.y * 128 + wm * 32;
    const int colBase = blockIdx.x * 128 + wn * 64;
    const int ntile0  = blockIdx.x * 8;      // 8 B tiles of 16 cols per block

    // ---- cooperative staging addresses (per thread: 32B of A, 32B of B) ----
    int agrow = blockIdx.y * 128 + (td >> 1);
    if (agrow > M - 1) agrow = M - 1;
    const __bf16* __restrict__ gA = A  + (size_t)agrow * K + (td & 1) * 16;
    const __bf16* __restrict__ gB = Bp + ((size_t)(ntile0 + (td >> 5)) * KT * 32
                                          + (td & 31)) * 16;
    const int ldsA_st = (td >> 1) * A_ROW_STRIDE + (td & 1) * 32;
    const int ldsB_st = A_TILE_BYTES + (td >> 5) * 1024 + (td & 31) * 32;

    // ---- per-wave fragment read offsets (bytes, within one buffer) ----
    const int ldsA_f0 = (wm * 32 + lh)      * A_ROW_STRIDE + khalfA * 2;
    const int ldsA_f1 = (wm * 32 + 16 + lh) * A_ROW_STRIDE + khalfA * 2;
    const int ldsB_f  = A_TILE_BYTES + (wn * 4) * 1024 + lane * 32;

    v8f acc[2][4] = {};

    // ---- prologue: stage K-step 0 into registers ----
    uint4 sa0 = *reinterpret_cast<const uint4*>(gA);
    uint4 sa1 = *reinterpret_cast<const uint4*>(gA + 8);
    uint4 sb0 = *reinterpret_cast<const uint4*>(gB);
    uint4 sb1 = *reinterpret_cast<const uint4*>(gB + 8);

    int it = 0;
    for (int kofs = 0; kofs < K; kofs += 32, it ^= 1) {
        unsigned char* buf = lds[it];
        // commit staged tile k to LDS
        *reinterpret_cast<uint4*>(buf + ldsA_st)      = sa0;
        *reinterpret_cast<uint4*>(buf + ldsA_st + 16) = sa1;
        *reinterpret_cast<uint4*>(buf + ldsB_st)      = sb0;
        *reinterpret_cast<uint4*>(buf + ldsB_st + 16) = sb1;
        // prefetch tile k+1 (global latency hidden behind this whole step)
        if (kofs + 32 < K) {
            const __bf16* nA = gA + (kofs + 32);
            const __bf16* nB = gB + (size_t)(kofs + 32) * 16;
            sa0 = *reinterpret_cast<const uint4*>(nA);
            sa1 = *reinterpret_cast<const uint4*>(nA + 8);
            sb0 = *reinterpret_cast<const uint4*>(nB);
            sb1 = *reinterpret_cast<const uint4*>(nB + 8);
        }
        __syncthreads();

        // fragment loads from LDS (ds_load_b128)
        Frag fa0, fa1, fb0, fb1, fb2, fb3;
        fa0.u[0] = *reinterpret_cast<const uint4*>(buf + ldsA_f0);
        fa0.u[1] = *reinterpret_cast<const uint4*>(buf + ldsA_f0 + 16);
        fa1.u[0] = *reinterpret_cast<const uint4*>(buf + ldsA_f1);
        fa1.u[1] = *reinterpret_cast<const uint4*>(buf + ldsA_f1 + 16);
        fb0.u[0] = *reinterpret_cast<const uint4*>(buf + ldsB_f);
        fb0.u[1] = *reinterpret_cast<const uint4*>(buf + ldsB_f + 16);
        fb1.u[0] = *reinterpret_cast<const uint4*>(buf + ldsB_f + 1024);
        fb1.u[1] = *reinterpret_cast<const uint4*>(buf + ldsB_f + 1040);
        fb2.u[0] = *reinterpret_cast<const uint4*>(buf + ldsB_f + 2048);
        fb2.u[1] = *reinterpret_cast<const uint4*>(buf + ldsB_f + 2064);
        fb3.u[0] = *reinterpret_cast<const uint4*>(buf + ldsB_f + 3072);
        fb3.u[1] = *reinterpret_cast<const uint4*>(buf + ldsB_f + 3088);

        acc[0][0] = WMMA_BF16(fa0.v, fb0.v, acc[0][0]);
        acc[1][0] = WMMA_BF16(fa1.v, fb0.v, acc[1][0]);
        acc[0][1] = WMMA_BF16(fa0.v, fb1.v, acc[0][1]);
        acc[1][1] = WMMA_BF16(fa1.v, fb1.v, acc[1][1]);
        acc[0][2] = WMMA_BF16(fa0.v, fb2.v, acc[0][2]);
        acc[1][2] = WMMA_BF16(fa1.v, fb2.v, acc[1][2]);
        acc[0][3] = WMMA_BF16(fa0.v, fb3.v, acc[0][3]);
        acc[1][3] = WMMA_BF16(fa1.v, fb3.v, acc[1][3]);
        // next iteration writes lds[it^1]; write-after-read on lds[it] is
        // protected by the barrier of the intervening iteration
    }

    // ---- store: lanes 0-15 -> cols, vgpr r -> row r / r+8 per half-wave
    const int rowSub = ((lane >> 4) << 3);
    const int col0   = colBase + lh;
    #pragma unroll
    for (int t = 0; t < 4; ++t) {
        const int col = col0 + t * 16;
        const float bv = bias ? bias[col] : 0.0f;
        #pragma unroll
        for (int ar = 0; ar < 2; ++ar) {
            const int rowTop = rowBase + ar * 16 + rowSub;
            #pragma unroll
            for (int r = 0; r < 8; ++r) {
                const int row = rowTop + r;
                if (row < M) {
                    float val = acc[ar][t][r] + bv;
                    if (relu) val = fmaxf(val, 0.0f);
                    const size_t idx = (size_t)row * Nc + col;
                    if (Cf) Cf[idx] = val;
                    if (Cb) Cb[idx] = (__bf16)val;
                }
            }
        }
    }
}

// ---------------------------------------------------------------------------
// score[e,h] = exp(clip( ((k[src]+rel) . q[dst]) / sqrt(DK), -10, 10 ))
// ---------------------------------------------------------------------------
__global__ __launch_bounds__(256)
void edge_score_kernel(const int* __restrict__ edges, const int* __restrict__ src,
                       const int* __restrict__ dst, const float* __restrict__ rel,
                       const float* __restrict__ q, const float* __restrict__ k,
                       float* __restrict__ score)
{
    int t = blockIdx.x * 256 + threadIdx.x;
    if (t >= N_EDGES * HEADS) return;
    const int e = t >> 3, h = t & 7;
    const int s = src[e], d = dst[e], r = edges[e];
    const float4* __restrict__ kp = (const float4*)(k + (size_t)s * DIM + h * DKH);
    const float4* __restrict__ qp = (const float4*)(q + (size_t)d * DIM + h * DKH);
    const float4* __restrict__ rp = (const float4*)(rel + r * DKH);
    float acc = 0.0f;
    #pragma unroll
    for (int j = 0; j < DKH / 4; ++j) {
        const float4 kv = kp[j], qv = qp[j], rv = rp[j];
        acc += (kv.x + rv.x) * qv.x + (kv.y + rv.y) * qv.y
             + (kv.z + rv.z) * qv.z + (kv.w + rv.w) * qv.w;
    }
    acc *= 0.125f;                              // 1/sqrt(64)
    acc = fminf(10.0f, fmaxf(-10.0f, acc));
    score[t] = expf(acc);
}

// ---------------------------------------------------------------------------
// wv[dst, d] += (v[src, d] + rel[edge, d%64]) * score[e, d/64]   (f32 atomics)
// z[dst, h]  += score[e, h]   (fused on dk==0 lanes)
// ---------------------------------------------------------------------------
__global__ __launch_bounds__(256)
void edge_scatter_kernel(const int* __restrict__ edges, const int* __restrict__ src,
                         const int* __restrict__ dst, const float* __restrict__ rel,
                         const float* __restrict__ v, const float* __restrict__ score,
                         float* __restrict__ wv, float* __restrict__ z)
{
    long long t = (long long)blockIdx.x * 256 + threadIdx.x;
    if (t >= (long long)N_EDGES * DIM) return;
    const int e  = (int)(t >> 9);
    const int d  = (int)(t & 511);
    const int h  = d >> 6;
    const int dk = d & 63;
    const float sc = score[e * HEADS + h];
    const float m  = (v[(size_t)src[e] * DIM + d] + rel[edges[e] * DKH + dk]) * sc;
    atomicAdd(&wv[(size_t)dst[e] * DIM + d], m);
    if (dk == 0) atomicAdd(&z[dst[e] * HEADS + h], sc);
}

// ---------------------------------------------------------------------------
// o(bf16)[n, d] = wv[n, d] / z[n, d/64]
// ---------------------------------------------------------------------------
__global__ __launch_bounds__(256)
void odiv_kernel(const float* __restrict__ wv, const float* __restrict__ z,
                 __bf16* __restrict__ ob)
{
    int t = blockIdx.x * 256 + threadIdx.x;
    if (t >= N_NODES * DIM) return;
    const int n = t >> 9;
    const int h = (t >> 6) & 7;
    ob[t] = (__bf16)(wv[t] / z[n * HEADS + h]);
}

// ---------------------------------------------------------------------------
// x = LayerNorm(x + proj) * g + b ; also emit bf16 copy. One block per node.
// ---------------------------------------------------------------------------
__global__ __launch_bounds__(256)
void ln_kernel(float* __restrict__ xcur, const float* __restrict__ proj,
               const float* __restrict__ g, const float* __restrict__ b,
               __bf16* __restrict__ xb)
{
    __shared__ float s1[256];
    __shared__ float s2[256];
    const int n = blockIdx.x;
    const int t = threadIdx.x;
    float* __restrict__ xp = xcur + (size_t)n * DIM;
    const float* __restrict__ pp = proj + (size_t)n * DIM;

    const float y0 = xp[t]       + pp[t];
    const float y1 = xp[t + 256] + pp[t + 256];
    s1[t] = y0 + y1;
    s2[t] = y0 * y0 + y1 * y1;
    __syncthreads();
    for (int s = 128; s > 0; s >>= 1) {
        if (t < s) { s1[t] += s1[t + s]; s2[t] += s2[t + s]; }
        __syncthreads();
    }
    const float mean = s1[0] * (1.0f / DIM);
    const float var  = s2[0] * (1.0f / DIM) - mean * mean;
    const float inv  = rsqrtf(var + LN_EPS);
    const float o0 = (y0 - mean) * inv * g[t]       + b[t];
    const float o1 = (y1 - mean) * inv * g[t + 256] + b[t + 256];
    xp[t]       = o0;
    xp[t + 256] = o1;
    xb[(size_t)n * DIM + t]       = (__bf16)o0;
    xb[(size_t)n * DIM + t + 256] = (__bf16)o1;
}

// ---------------------------------------------------------------------------
// launch
// ---------------------------------------------------------------------------
extern "C" void kernel_launch(void* const* d_in, const int* in_sizes, int n_in,
                              void* d_out, int out_size, void* d_ws, size_t ws_size,
                              hipStream_t stream)
{
    (void)in_sizes; (void)n_in; (void)out_size; (void)ws_size;

    const float* x    = (const float*)d_in[0];
    const int*   edges= (const int*)d_in[1];
    const int*   srcA = (const int*)d_in[2];
    const int*   dstA = (const int*)d_in[3];
    const float* rel  = (const float*)d_in[4];
    const float* Wq   = (const float*)d_in[5];
    const float* bq   = (const float*)d_in[6];
    const float* Wk   = (const float*)d_in[7];
    const float* Wv   = (const float*)d_in[8];
    const float* Wo   = (const float*)d_in[9];
    const float* bo   = (const float*)d_in[10];
    const float* ln1g = (const float*)d_in[11];
    const float* ln1b = (const float*)d_in[12];
    const float* W1   = (const float*)d_in[13];
    const float* b1   = (const float*)d_in[14];
    const float* W2   = (const float*)d_in[15];
    const float* b2   = (const float*)d_in[16];
    const float* ln2g = (const float*)d_in[17];
    const float* ln2b = (const float*)d_in[18];

    // ---- workspace arena ----
    char* p = (char*)d_ws;
    auto alloc = [&](size_t bytes) -> char* {
        char* r = p;
        p += (bytes + 255) & ~(size_t)255;
        return r;
    };
    const size_t DD  = (size_t)DIM * DIM;        // 262144
    const size_t D4D = (size_t)DIM * 4 * DIM;    // 1048576
    __bf16* WqB[LAYERS]; __bf16* WkB[LAYERS]; __bf16* WvB[LAYERS];
    __bf16* WoB[LAYERS]; __bf16* W1B[LAYERS]; __bf16* W2B[LAYERS];
    for (int i = 0; i < LAYERS; ++i) {
        WqB[i] = (__bf16*)alloc(DD  * 2);
        WkB[i] = (__bf16*)alloc(DD  * 2);
        WvB[i] = (__bf16*)alloc(DD  * 2);
        WoB[i] = (__bf16*)alloc(DD  * 2);
        W1B[i] = (__bf16*)alloc(D4D * 2);
        W2B[i] = (__bf16*)alloc(D4D * 2);
    }
    const size_t ND = (size_t)N_NODES * DIM;     // 10,240,000
    float*  xcur  = (float*) alloc(ND * 4);
    __bf16* xb    = (__bf16*)alloc(ND * 2);
    float*  qbuf  = (float*) alloc(ND * 4);
    float*  kbuf  = (float*) alloc(ND * 4);      // contiguous with qbuf
    float*  vbuf  = (float*) alloc(ND * 4);
    float*  score = (float*) alloc((size_t)N_EDGES * HEADS * 4);
    float*  zbuf  = (float*) alloc((size_t)N_NODES * HEADS * 4);
    float*  wv    = (float*) alloc(ND * 4);
    __bf16* ob    = (__bf16*)alloc(ND * 2);
    float*  proj  = (float*) alloc(ND * 4);
    __bf16* hb    = (__bf16*)qbuf;               // alias: q+k region (82 MB) holds FFN hidden

    // ---- weight packing (f32 -> bf16 WMMA-B fragment layout) ----
    const int gDD  = (int)((DD  + 255) / 256);
    const int gD4D = (int)((D4D + 255) / 256);
    for (int i = 0; i < LAYERS; ++i) {
        pack_b_kernel<<<gDD,  256, 0, stream>>>(Wq + i * DD,  WqB[i], DIM,     DIM);
        pack_b_kernel<<<gDD,  256, 0, stream>>>(Wk + i * DD,  WkB[i], DIM,     DIM);
        pack_b_kernel<<<gDD,  256, 0, stream>>>(Wv + i * DD,  WvB[i], DIM,     DIM);
        pack_b_kernel<<<gDD,  256, 0, stream>>>(Wo + i * DD,  WoB[i], DIM,     DIM);
        pack_b_kernel<<<gD4D, 256, 0, stream>>>(W1 + i * D4D, W1B[i], DIM,     4 * DIM);
        pack_b_kernel<<<gD4D, 256, 0, stream>>>(W2 + i * D4D, W2B[i], 4 * DIM, DIM);
    }

    // ---- x init ----
    hipMemcpyAsync(xcur, x, ND * 4, hipMemcpyDeviceToDevice, stream);
    f2bf_kernel<<<(int)((ND + 255) / 256), 256, 0, stream>>>(x, xb, (int)ND);

    const dim3 blk(256);
    const int  mblocks = (N_NODES + 127) / 128;          // 157
    const dim3 gD (DIM / 128,     mblocks);              // Nc=512  -> x=4
    const dim3 g4D(4 * DIM / 128, mblocks);              // Nc=2048 -> x=16

    for (int i = 0; i < LAYERS; ++i) {
        // q / k / v projections
        gemm_wmma_bf16_t<DIM><<<gD, blk, 0, stream>>>(xb, WqB[i], bq + (size_t)i * DIM,
                                                      qbuf, nullptr, N_NODES, DIM, 0);
        gemm_wmma_bf16_t<DIM><<<gD, blk, 0, stream>>>(xb, WkB[i], nullptr,
                                                      kbuf, nullptr, N_NODES, DIM, 0);
        gemm_wmma_bf16_t<DIM><<<gD, blk, 0, stream>>>(xb, WvB[i], nullptr,
                                                      vbuf, nullptr, N_NODES, DIM, 0);

        // attention scores
        edge_score_kernel<<<(N_EDGES * HEADS + 255) / 256, blk, 0, stream>>>(
            edges, srcA, dstA, rel, qbuf, kbuf, score);

        // scatter messages
        hipMemsetAsync(wv,   0, ND * 4, stream);
        hipMemsetAsync(zbuf, 0, (size_t)N_NODES * HEADS * 4, stream);
        edge_scatter_kernel<<<(int)(((long long)N_EDGES * DIM + 255) / 256), blk, 0, stream>>>(
            edges, srcA, dstA, rel, vbuf, score, wv, zbuf);

        // o = wv / z  (bf16 for Wo GEMM)
        odiv_kernel<<<(N_NODES * DIM + 255) / 256, blk, 0, stream>>>(wv, zbuf, ob);

        // attention output projection + residual layernorm
        gemm_wmma_bf16_t<DIM><<<gD, blk, 0, stream>>>(ob, WoB[i], bo + (size_t)i * DIM,
                                                      proj, nullptr, N_NODES, DIM, 0);
        ln_kernel<<<N_NODES, blk, 0, stream>>>(xcur, proj,
                                               ln1g + (size_t)i * DIM,
                                               ln1b + (size_t)i * DIM, xb);

        // FFN
        gemm_wmma_bf16_t<DIM><<<g4D, blk, 0, stream>>>(xb, W1B[i], b1 + (size_t)i * 4 * DIM,
                                                       nullptr, hb, N_NODES, 4 * DIM, 1);
        gemm_wmma_bf16_t<4 * DIM><<<gD, blk, 0, stream>>>(hb, W2B[i], b2 + (size_t)i * DIM,
                                                          proj, nullptr, N_NODES, DIM, 0);
        ln_kernel<<<N_NODES, blk, 0, stream>>>(xcur, proj,
                                               ln2g + (size_t)i * DIM,
                                               ln2b + (size_t)i * DIM, xb);
    }

    hipMemcpyAsync(d_out, xcur, ND * 4, hipMemcpyDeviceToDevice, stream);
}
